// StochSelfAttention_56349970923833
// MI455X (gfx1250) — compile-verified, matrix operand order checked by reading, not run on hardware
//
#include <hip/hip_runtime.h>
#include <hip/hip_bf16.h>
#include <stdint.h>

// ---------------------------------------------------------------------------
// StochSelfAttention on gfx1250 (MI455X).
// Pipeline:
//   1. convert_x : x fp32 -> bf16 (one pass, halves A-read traffic)
//   2. convert_w : pack Wq|Wk|Wv -> bf16 [2304][768], biases, W1 -> bf16 [64][1536]
//   3. qkv_gemm  : bf16 WMMA GEMM, M=50432 K=768 N=2304 -> Q,K,V fp32 (+bias)
//                  A/B tiles staged with GLOBAL_LOAD_ASYNC_TO_LDS_B128,
//                  double-buffered, synced with s_wait_asynccnt (ASYNCcnt).
//   4. mlp1      : bf16 WMMA GEMM, A=concat(q_nc,k_nc) M=50176 K=1536 N=64
//   5. mlp23     : scalar 60x60 + 60x2 layers + sigmoid -> sampling grid
//   6. finalize  : bilinear grid-sample of K/V + sigmoid(0.01*<sk,q>)*sv
// ---------------------------------------------------------------------------

typedef __attribute__((ext_vector_type(16))) __bf16 bf16x16;
typedef __attribute__((ext_vector_type(8)))  float  f32x8;

#define TEMPF 0.01f
#define BATCH 256
#define SEQ   197
#define DIM   768
#define GG    14
#define PP    196               // GG*GG
#define MQKV  (BATCH*SEQ)       // 50432 = 394*128
#define NQKV  (3*DIM)           // 2304  = 36*64
#define MMLP  (BATCH*PP)        // 50176 = 392*128
#define KMLP  (2*DIM)           // 1536
#define H60   60

static __device__ __forceinline__ uint16_t f2bf(float f) {
  union { float f; uint32_t u; } x; x.f = f;
  uint32_t r = x.u + 0x7FFFu + ((x.u >> 16) & 1u);   // round-to-nearest-even
  return (uint16_t)(r >> 16);
}

// K-index base for 16-bit A/B fragment register r, lane-half h (ISA 7.12.2,
// 16-bit A-matrix 16x32: VGPR0..3 hold K={0..7 | 8..15}, VGPR4..7 hold +16).
static __device__ __forceinline__ int kmap(int r, int h) {
  return (r < 4) ? (h * 8 + 2 * r) : (16 + h * 8 + 2 * (r - 4));
}

// Generic pointer to a __shared__ object is {LDS aperture, ds_offset32};
// the low 32 bits are the byte offset the async-DMA VDST operand wants.
static __device__ __forceinline__ uint32_t lds_off(const void* p) {
  return (uint32_t)(uintptr_t)p;
}

// ---------------------------------------------------------------------------
__global__ void convert_x_kernel(const float* __restrict__ x,
                                 uint16_t* __restrict__ xb, int n4) {
  int i = blockIdx.x * blockDim.x + threadIdx.x;
  if (i >= n4) return;
  float4 v = ((const float4*)x)[i];
  uint32_t p0 = (uint32_t)f2bf(v.x) | ((uint32_t)f2bf(v.y) << 16);
  uint32_t p1 = (uint32_t)f2bf(v.z) | ((uint32_t)f2bf(v.w) << 16);
  ((uint2*)xb)[i] = make_uint2(p0, p1);
}

__global__ void convert_w_kernel(const float* __restrict__ Wq, const float* __restrict__ Wk,
                                 const float* __restrict__ Wv, const float* __restrict__ bq,
                                 const float* __restrict__ bk, const float* __restrict__ bv,
                                 const float* __restrict__ W1,
                                 uint16_t* __restrict__ wqkvb, float* __restrict__ bqkv,
                                 uint16_t* __restrict__ w1b) {
  int i = blockIdx.x * blockDim.x + threadIdx.x;
  if (i < NQKV * DIM) {
    int n = i / DIM, k = i - n * DIM;
    float v = (n < DIM) ? Wq[n * DIM + k]
            : (n < 2 * DIM) ? Wk[(n - DIM) * DIM + k]
            : Wv[(n - 2 * DIM) * DIM + k];
    wqkvb[i] = f2bf(v);
    return;
  }
  i -= NQKV * DIM;
  if (i < NQKV) {
    bqkv[i] = (i < DIM) ? bq[i] : (i < 2 * DIM) ? bk[i - DIM] : bv[i - 2 * DIM];
    return;
  }
  i -= NQKV;
  if (i < 64 * KMLP) {
    int f = i / KMLP, k = i - f * KMLP;
    w1b[i] = (f < H60) ? f2bf(W1[f * KMLP + k]) : (uint16_t)0;
  }
}

// ---------------------------------------------------------------------------
// Fused QKV projection: C[m, n] = sum_k xb[m,k] * Wqkv[n,k]  (+ bias)
// Block tile 128(M) x 64(N), K-step 32 (one WMMA K). 8 waves = 4x2, each wave
// owns a 32x32 quadrant (2x2 WMMA tiles). Tiles staged via async DMA to LDS.
#define ASTR 40   // row stride in shorts: 80B rows -> 16B-aligned for b128 DMA,
                  // 20-bank stride -> 16 consecutive rows hit 16 distinct banks.
__global__ __launch_bounds__(256) void qkv_gemm_kernel(
    const uint16_t* __restrict__ Xb, const uint16_t* __restrict__ Wb,
    const float* __restrict__ bias,
    float* __restrict__ Q, float* __restrict__ K, float* __restrict__ V) {
  __shared__ uint16_t As[2][128 * ASTR];
  __shared__ uint16_t Bs[2][64 * ASTR];
  const int tid  = threadIdx.x;
  const int lane = tid & 31, wave = tid >> 5;
  const int wm = wave & 3, wn = wave >> 2;
  const int half = lane >> 4, l16 = lane & 15;
  const size_t m0 = (size_t)blockIdx.y * 128;
  const int    n0 = blockIdx.x * 64;

  // Issue one tile's async global->LDS copies (3 wave-ops: 2 for A, 1 for B).
  auto issueTile = [&](int bufIdx, int k0) {
#pragma unroll
    for (int it = 0; it < 2; ++it) {           // A tile: 128x32 bf16 (8 KB)
      int i = tid + it * 256;
      int row = i >> 2, kq = (i & 3) << 3;
      uint64_t g = (uint64_t)(uintptr_t)(Xb + (m0 + row) * DIM + k0 + kq);
      uint32_t l = lds_off(&As[bufIdx][row * ASTR + kq]);
      asm volatile("global_load_async_to_lds_b128 %0, %1, off"
                   :: "v"(l), "v"(g) : "memory");
    }
    {                                          // B tile: 64x32 bf16 (4 KB)
      int row = tid >> 2, kq = (tid & 3) << 3;
      uint64_t g = (uint64_t)(uintptr_t)(Wb + (size_t)(n0 + row) * DIM + k0 + kq);
      uint32_t l = lds_off(&Bs[bufIdx][row * ASTR + kq]);
      asm volatile("global_load_async_to_lds_b128 %0, %1, off"
                   :: "v"(l), "v"(g) : "memory");
    }
  };

  f32x8 acc[2][2] = {};
  const int NT = DIM / 32;                     // 24 K-tiles
  issueTile(0, 0);
  for (int t = 0; t < NT; ++t) {
    const int cur = t & 1;
    if (t + 1 < NT) {
      __builtin_prefetch(Xb + (m0 + (tid >> 1)) * DIM + t * 32 + 64, 0, 1);
      issueTile(cur ^ 1, (t + 1) * 32);
      // 3 ops just issued for tile t+1 may remain; tile t's (in-order) are done.
      asm volatile("s_wait_asynccnt 3" ::: "memory");
    } else {
      asm volatile("s_wait_asynccnt 0" ::: "memory");
    }
    __syncthreads();                           // buf[cur] loaded by all waves

    union FB { uint32_t u[8]; bf16x16 v; };
    FB bfr[2];
#pragma unroll
    for (int j = 0; j < 2; ++j) {
      int nr = wn * 32 + j * 16 + l16;
#pragma unroll
      for (int r = 0; r < 8; ++r)
        bfr[j].u[r] = *(const uint32_t*)&Bs[cur][nr * ASTR + kmap(r, half)];
    }
#pragma unroll
    for (int i = 0; i < 2; ++i) {
      int mr = wm * 32 + i * 16 + l16;
      FB af;
#pragma unroll
      for (int r = 0; r < 8; ++r)
        af.u[r] = *(const uint32_t*)&As[cur][mr * ASTR + kmap(r, half)];
#pragma unroll
      for (int j = 0; j < 2; ++j)
        acc[i][j] = __builtin_amdgcn_wmma_f32_16x16x32_bf16(
            false, af.v, false, bfr[j].v, (short)0, acc[i][j], false, false);
    }
    __syncthreads();                           // all reads of buf[cur] done
  }
#pragma unroll
  for (int i = 0; i < 2; ++i)
#pragma unroll
    for (int j = 0; j < 2; ++j) {
      int nG = n0 + wn * 32 + j * 16 + l16;
      float bsv = bias[nG];
      int w = nG / DIM;
      int col = nG - w * DIM;
      float* out = (w == 0) ? Q : (w == 1) ? K : V;
#pragma unroll
      for (int e = 0; e < 8; ++e) {            // C layout: m = e + 8*half, n = l16
        size_t mG = m0 + wm * 32 + i * 16 + e + half * 8;
        out[mG * DIM + col] = acc[i][j][e] + bsv;
      }
    }
}

// ---------------------------------------------------------------------------
// MLP layer 1 as WMMA GEMM: A = concat(q_nc, k_nc) (fp32 -> bf16 in flight
// while staging LDS, so classic ds_store staging), B = W1 padded to 64 rows.
__global__ __launch_bounds__(256) void mlp1_kernel(
    const float* __restrict__ Q, const float* __restrict__ K,
    const uint16_t* __restrict__ W1b, const float* __restrict__ b1,
    float* __restrict__ H1out) {
  __shared__ uint16_t As[128 * 36];
  __shared__ uint16_t Bs[64 * 36];
  const int tid  = threadIdx.x;
  const int lane = tid & 31, wave = tid >> 5;
  const int wm = wave & 3, wn = wave >> 2;
  const int half = lane >> 4, l16 = lane & 15;
  const int m0 = blockIdx.y * 128;
  f32x8 acc[2][2] = {};
  for (int k0 = 0; k0 < KMLP; k0 += 32) {
#pragma unroll
    for (int it = 0; it < 2; ++it) {
      int i = tid + it * 256;
      int row = i >> 2, kq = (i & 3) << 3;
      int m = m0 + row;
      int b = m / PP;
      int s = 1 + (m - b * PP);               // skip cls token
      size_t base = ((size_t)b * SEQ + s) * DIM;
      int gk = k0 + kq;                        // 768 boundary is 8-aligned
      const float* src = (gk < DIM) ? (Q + base + gk) : (K + base + gk - DIM);
      float4 f0 = *(const float4*)src;
      float4 f1 = *(const float4*)(src + 4);
      uint32_t* dst = (uint32_t*)&As[row * 36 + kq];
      dst[0] = (uint32_t)f2bf(f0.x) | ((uint32_t)f2bf(f0.y) << 16);
      dst[1] = (uint32_t)f2bf(f0.z) | ((uint32_t)f2bf(f0.w) << 16);
      dst[2] = (uint32_t)f2bf(f1.x) | ((uint32_t)f2bf(f1.y) << 16);
      dst[3] = (uint32_t)f2bf(f1.z) | ((uint32_t)f2bf(f1.w) << 16);
    }
    {
      int row = tid >> 2, kq = (tid & 3) << 3;
      uint4 d = *(const uint4*)(W1b + (size_t)row * KMLP + k0 + kq);
      uint32_t* dst = (uint32_t*)&Bs[row * 36 + kq];
      dst[0] = d.x; dst[1] = d.y; dst[2] = d.z; dst[3] = d.w;
    }
    __syncthreads();
    union FB { uint32_t u[8]; bf16x16 v; };
    FB bfr[2];
#pragma unroll
    for (int j = 0; j < 2; ++j) {
      int nr = wn * 32 + j * 16 + l16;
#pragma unroll
      for (int r = 0; r < 8; ++r)
        bfr[j].u[r] = *(const uint32_t*)&Bs[nr * 36 + kmap(r, half)];
    }
#pragma unroll
    for (int i = 0; i < 2; ++i) {
      int mr = wm * 32 + i * 16 + l16;
      FB af;
#pragma unroll
      for (int r = 0; r < 8; ++r)
        af.u[r] = *(const uint32_t*)&As[mr * 36 + kmap(r, half)];
#pragma unroll
      for (int j = 0; j < 2; ++j)
        acc[i][j] = __builtin_amdgcn_wmma_f32_16x16x32_bf16(
            false, af.v, false, bfr[j].v, (short)0, acc[i][j], false, false);
    }
    __syncthreads();
  }
#pragma unroll
  for (int i = 0; i < 2; ++i)
#pragma unroll
    for (int j = 0; j < 2; ++j) {
      int nG = wn * 32 + j * 16 + l16;         // 0..63 (single N tile column)
      float bsv = (nG < H60) ? b1[nG] : 0.f;
#pragma unroll
      for (int e = 0; e < 8; ++e) {
        size_t mG = (size_t)m0 + wm * 32 + i * 16 + e + half * 8;
        H1out[mG * 64 + nG] = fmaxf(acc[i][j][e] + bsv, 0.f);
      }
    }
}

// ---------------------------------------------------------------------------
__global__ __launch_bounds__(256) void mlp23_kernel(
    const float* __restrict__ H1, const float* __restrict__ W2,
    const float* __restrict__ b2, const float* __restrict__ W3,
    const float* __restrict__ b3, float* __restrict__ sample) {
  __shared__ float sW2[H60 * H60];
  __shared__ float sW3[2 * H60];
  __shared__ float sH[32 * H60];
  __shared__ float sH2[32 * H60];
  const int tid = threadIdx.x;
  const int r0 = blockIdx.x * 32;
  for (int i = tid; i < H60 * H60; i += 256) sW2[i] = W2[i];
  for (int i = tid; i < 2 * H60; i += 256) sW3[i] = W3[i];
  for (int i = tid; i < 32 * H60; i += 256) {
    int r = i / H60, f = i - r * H60;
    sH[i] = H1[(size_t)(r0 + r) * 64 + f];
  }
  __syncthreads();
  for (int i = tid; i < 32 * H60; i += 256) {
    int r = i / H60, f = i - r * H60;
    float a = b2[f];
#pragma unroll 4
    for (int k = 0; k < H60; ++k) a += sH[r * H60 + k] * sW2[f * H60 + k];
    sH2[i] = fmaxf(a, 0.f);
  }
  __syncthreads();
  if (tid < 64) {
    int r = tid >> 1, c = tid & 1;
    float a = b3[c];
#pragma unroll 4
    for (int k = 0; k < H60; ++k) a += sH2[r * H60 + k] * sW3[c * H60 + k];
    sample[(size_t)(r0 + r) * 2 + c] = 1.f / (1.f + __expf(-a));
  }
}

// ---------------------------------------------------------------------------
// One block per (b, s): bilinear-sample k/v, reduce <sk,q> over D=768,
// out = sigmoid(TEMP*score) * sv.  cls row (s==0): sk=sv=1.
__global__ __launch_bounds__(256) void finalize_kernel(
    const float* __restrict__ Q, const float* __restrict__ K,
    const float* __restrict__ V, const float* __restrict__ sample,
    float* __restrict__ out) {
  const int bs = blockIdx.x;
  const int b = bs / SEQ, s = bs - b * SEQ;
  const int tid = threadIdx.x;
  const int lane = tid & 31, wave = tid >> 5;
  __shared__ float red[8];
  __shared__ float s_a;

  float  w4[4] = {0.f, 0.f, 0.f, 0.f};
  size_t cb[4] = {0, 0, 0, 0};
  const bool cls = (s == 0);
  if (!cls) {
    int p = s - 1;
    int fx = 2 * p, fy = 2 * p + 1;
    // grid = concat(sample_x, sample_y, axis=1).reshape(B,14,14,2)
    float gx = (fx < PP) ? sample[((size_t)b * PP + fx) * 2 + 0]
                         : sample[((size_t)b * PP + fx - PP) * 2 + 1];
    float gy = (fy < PP) ? sample[((size_t)b * PP + fy) * 2 + 0]
                         : sample[((size_t)b * PP + fy - PP) * 2 + 1];
    float ix = ((gx + 1.f) * (float)GG - 1.f) * 0.5f;
    float iy = ((gy + 1.f) * (float)GG - 1.f) * 0.5f;
    float ix0f = floorf(ix), iy0f = floorf(iy);
    int ix0 = (int)ix0f, iy0 = (int)iy0f;
    float wx1 = ix - ix0f, wx0 = 1.f - wx1;
    float wy1 = iy - iy0f, wy0 = 1.f - wy1;
    const int   xs[4] = {ix0, ix0 + 1, ix0, ix0 + 1};
    const int   ys[4] = {iy0, iy0, iy0 + 1, iy0 + 1};
    const float ws[4] = {wy0 * wx0, wy0 * wx1, wy1 * wx0, wy1 * wx1};
#pragma unroll
    for (int c = 0; c < 4; ++c) {
      bool valid = (xs[c] >= 0) && (xs[c] <= GG - 1) &&
                   (ys[c] >= 0) && (ys[c] <= GG - 1);
      w4[c] = valid ? ws[c] : 0.f;
      int xc = min(max(xs[c], 0), GG - 1), yc = min(max(ys[c], 0), GG - 1);
      cb[c] = ((size_t)b * SEQ + 1 + yc * GG + xc) * DIM;
    }
  }
  const size_t qbase = ((size_t)b * SEQ + s) * DIM;
  float qv[3], skv[3], svv[3];
#pragma unroll
  for (int t = 0; t < 3; ++t) {
    int d = tid + t * 256;
    qv[t] = Q[qbase + d];
    if (cls) {
      skv[t] = 1.f; svv[t] = 1.f;
    } else {
      float sk = 0.f, sv = 0.f;
#pragma unroll
      for (int c = 0; c < 4; ++c) {
        sk += w4[c] * K[cb[c] + d];
        sv += w4[c] * V[cb[c] + d];
      }
      skv[t] = sk; svv[t] = sv;
    }
  }
  float partial = qv[0] * skv[0] + qv[1] * skv[1] + qv[2] * skv[2];
#pragma unroll
  for (int o = 16; o > 0; o >>= 1) partial += __shfl_down(partial, o, 32);
  if (lane == 0) red[wave] = partial;
  __syncthreads();
  if (tid == 0) {
    float tot = 0.f;
#pragma unroll
    for (int wv = 0; wv < 8; ++wv) tot += red[wv];
    s_a = 1.f / (1.f + __expf(-TEMPF * tot));
  }
  __syncthreads();
  const float a = s_a;
#pragma unroll
  for (int t = 0; t < 3; ++t) out[qbase + tid + t * 256] = a * svv[t];
}

// ---------------------------------------------------------------------------
extern "C" void kernel_launch(void* const* d_in, const int* in_sizes, int n_in,
                              void* d_out, int out_size, void* d_ws, size_t ws_size,
                              hipStream_t stream) {
  (void)in_sizes; (void)n_in; (void)out_size; (void)ws_size;
  const float* x  = (const float*)d_in[0];
  // d_in[1] = mask (unused by the reference)
  const float* Wq = (const float*)d_in[2];
  const float* bq = (const float*)d_in[3];
  const float* Wk = (const float*)d_in[4];
  const float* bk = (const float*)d_in[5];
  const float* Wv = (const float*)d_in[6];
  const float* bv = (const float*)d_in[7];
  const float* W1 = (const float*)d_in[8];
  const float* b1 = (const float*)d_in[9];
  const float* W2 = (const float*)d_in[10];
  const float* b2 = (const float*)d_in[11];
  const float* W3 = (const float*)d_in[12];
  const float* b3 = (const float*)d_in[13];
  float* out = (float*)d_out;

  char* base = (char*)d_ws;
  size_t off = 0;
  auto take = [&](size_t bytes) {
    char* p = base + off;
    off = (off + bytes + 255) & ~(size_t)255;
    return p;
  };
  uint16_t* xb    = (uint16_t*)take((size_t)MQKV * DIM * 2);   // x in bf16
  uint16_t* wqkvb = (uint16_t*)take((size_t)NQKV * DIM * 2);   // Wq|Wk|Wv bf16
  float*    bqkv  = (float*)   take((size_t)NQKV * 4);
  uint16_t* w1b   = (uint16_t*)take((size_t)64 * KMLP * 2);    // W1 padded bf16
  float*    Qb    = (float*)   take((size_t)MQKV * DIM * 4);
  float*    Kb    = (float*)   take((size_t)MQKV * DIM * 4);
  float*    Vb    = (float*)   take((size_t)MQKV * DIM * 4);
  float*    H1    = (float*)   take((size_t)MMLP * 64 * 4);
  float*    smp   = (float*)   take((size_t)MMLP * 2 * 4);

  const int n4 = MQKV * DIM / 4;
  convert_x_kernel<<<(n4 + 255) / 256, 256, 0, stream>>>(x, xb, n4);
  const int wtot = NQKV * DIM + NQKV + 64 * KMLP;
  convert_w_kernel<<<(wtot + 255) / 256, 256, 0, stream>>>(
      Wq, Wk, Wv, bq, bk, bv, W1, wqkvb, bqkv, w1b);
  qkv_gemm_kernel<<<dim3(NQKV / 64, MQKV / 128), 256, 0, stream>>>(
      xb, wqkvb, bqkv, Qb, Kb, Vb);
  mlp1_kernel<<<dim3(1, MMLP / 128), 256, 0, stream>>>(Qb, Kb, w1b, b1, H1);
  mlp23_kernel<<<MMLP / 32, 256, 0, stream>>>(H1, W2, b2, W3, b3, smp);
  finalize_kernel<<<MQKV, 256, 0, stream>>>(Qb, Kb, Vb, smp, out);
}
